// Deepseekv3MoE_87900800680232
// MI455X (gfx1250) — matrix operand order repaired and checked by reading, not compiled, not run on hardware
//
#include <hip/hip_runtime.h>

// ---------------------------------------------------------------------------
// Problem constants (DeepSeek-V3 MoE layer)
// ---------------------------------------------------------------------------
#define T_ 4096
#define H_ 1024
#define E_ 64
#define I_ 512
#define IS_ 2048
#define NGRP_ 8
#define TOPKG_ 4
#define TOPK_ 8
#define RSCALE_ 2.5f

typedef unsigned short u16;
typedef __attribute__((ext_vector_type(16))) __bf16 v16bf;
typedef __attribute__((ext_vector_type(8)))  float  v8f;
typedef __attribute__((ext_vector_type(4)))  unsigned v4u;
typedef __attribute__((ext_vector_type(8)))  int      v8i;
typedef __attribute__((ext_vector_type(4)))  int      v4i;

union FragU { v16bf v; uint4 q[2]; };

static __device__ __forceinline__ u16 f32_to_bf16(float f) {
  unsigned u = __float_as_uint(f);
  unsigned r = (u + 0x7FFFu + ((u >> 16) & 1u)) >> 16;
  return (u16)r;
}

// A-fragment (16x32 bf16, MxK). Layout per CDNA5 ISA 7.12.2:
// lanes 0-15 row=lane K={0..7,16..23}; lanes 16-31 row=lane-16 K={8..15,24..31}.
static __device__ __forceinline__ v16bf load_frag_a(const u16* s, int stride) {
  int lane = threadIdx.x & 31;
  int row  = lane & 15;
  int k0   = (lane >> 4) << 3;
  const u16* p = s + row * stride;
  FragU f;
  f.q[0] = *(const uint4*)(p + k0);
  f.q[1] = *(const uint4*)(p + 16 + k0);
  return f.v;
}

// B-fragment (32x16, KxN), staged as [N rows][K contiguous].
// Lanes 0-15: col=lane K={0..7,16..23}; lanes 16-31: col=lane-16 K={8..15,24..31}.
static __device__ __forceinline__ v16bf load_frag_b(const u16* s, int stride) {
  int lane = threadIdx.x & 31;
  int col  = lane & 15;
  int k0   = (lane >> 4) << 3;
  const u16* p = s + col * stride + k0;
  FragU f;
  f.q[0] = *(const uint4*)(p);
  f.q[1] = *(const uint4*)(p + 16);
  return f.v;
}

static __device__ __forceinline__ v8f wmma_bf16(v16bf a, v16bf b, v8f c) {
  return __builtin_amdgcn_wmma_f32_16x16x32_bf16(false, a, false, b,
                                                 (short)0, c, false, false);
}

static __device__ __forceinline__ float silu(float x) {
  return x / (1.0f + __expf(-x));
}

// ---------------------------------------------------------------------------
// TDM: async 2D tile load (rows x 64 bf16 elements) global -> LDS.
// LDS destination gets 4 DWORDs of padding after each 32-DWORD (64 elem) row,
// i.e. the 72-u16 row stride the WMMA fragment loaders expect.
// D# layout per CDNA5 ISA.
// ---------------------------------------------------------------------------
static __device__ __forceinline__ void tdm_load_tile_2d(
    void* lds_dst, const u16* gsrc, int rows, int stride_elems) {
  unsigned lds_addr = (unsigned)(uintptr_t)lds_dst;   // low 32 bits of flat LDS addr = LDS offset
  unsigned long long ga = (unsigned long long)(uintptr_t)gsrc;
  v4u g0;
  g0[0] = 1u;                                          // count=1, user descriptor
  g0[1] = lds_addr;                                    // lds_addr
  g0[2] = (unsigned)ga;                                // global_addr[31:0]
  g0[3] = (unsigned)((ga >> 32) & 0x1FFFFFFu) | 0x80000000u;  // global_addr[56:32], type=2
  v8i g1;
  g1[0] = (int)((1u << 16)      // data_size = 2 bytes
              | (1u << 20)      // pad_enable
              | (4u << 22)      // pad_interval: 32 DWORDs (one 64-elem row)
              | (3u << 25));    // pad_amount: 4 DWORDs (16B)
  g1[1] = (int)((64u & 0xFFFFu) << 16);                // tensor_dim0 lo16 = 64
  g1[2] = (int)(((unsigned)rows & 0xFFFFu) << 16);     // dim0 hi=0, tensor_dim1 lo16 = rows
  g1[3] = (int)(64u << 16);                            // dim1 hi=0, tile_dim0 = 64
  g1[4] = (int)((unsigned)rows & 0xFFFFu);             // tile_dim1 = rows, tile_dim2 = 0
  g1[5] = stride_elems;                                // tensor_dim0_stride lo32
  g1[6] = 0;
  g1[7] = 0;
  v4i z4 = {0, 0, 0, 0};
#if __clang_major__ >= 23
  v8i z8 = {0, 0, 0, 0, 0, 0, 0, 0};
  __builtin_amdgcn_tensor_load_to_lds(g0, g1, z4, z4, z8, 0);
#else
  __builtin_amdgcn_tensor_load_to_lds(g0, g1, z4, z4, 0);
#endif
}

// ---------------------------------------------------------------------------
// fp32 -> bf16 conversion
// ---------------------------------------------------------------------------
__global__ void cvt_f32_bf16(const float* __restrict__ in,
                             uint2* __restrict__ out, int n4) {
  int i = blockIdx.x * blockDim.x + threadIdx.x;
  if (i >= n4) return;
  float4 v = ((const float4*)in)[i];
  uint2 o;
  o.x = (unsigned)f32_to_bf16(v.x) | ((unsigned)f32_to_bf16(v.y) << 16);
  o.y = (unsigned)f32_to_bf16(v.z) | ((unsigned)f32_to_bf16(v.w) << 16);
  out[i] = o;
}

// ---------------------------------------------------------------------------
// Router: 1 block / token, thread e = expert e.
// ---------------------------------------------------------------------------
__global__ __launch_bounds__(E_) void router_kernel(
    const float* __restrict__ x, const float* __restrict__ gw,
    const float* __restrict__ ebias, int* __restrict__ counts,
    int* __restrict__ tok, float* __restrict__ wt) {
  __shared__ float sc[E_];
  __shared__ float swb[E_];
  const int t = blockIdx.x;
  const int e = threadIdx.x;

  const float4* xr = (const float4*)(x + (size_t)t * H_);
  const float4* gr = (const float4*)(gw + (size_t)e * H_);
  float acc = 0.0f;
#pragma unroll 8
  for (int i = 0; i < H_ / 4; ++i) {
    float4 a = xr[i], b = gr[i];
    acc += a.x * b.x + a.y * b.y + a.z * b.z + a.w * b.w;
  }
  float s = 1.0f / (1.0f + __expf(-acc));
  sc[e]  = s;
  swb[e] = s + ebias[e];
  __syncthreads();

  if (e == 0) {
    float gs[NGRP_];
    for (int g = 0; g < NGRP_; ++g) {
      float m1 = -1e30f, m2 = -1e30f;
      for (int j = 0; j < E_ / NGRP_; ++j) {
        float v = swb[g * (E_ / NGRP_) + j];
        if (v > m1) { m2 = m1; m1 = v; }
        else if (v > m2) { m2 = v; }
      }
      gs[g] = m1 + m2;
    }
    unsigned gmask = 0;
    for (int k = 0; k < TOPKG_; ++k) {
      int best = 0; float bv = -1e30f;
      for (int g = 0; g < NGRP_; ++g)
        if (!((gmask >> g) & 1u) && gs[g] > bv) { bv = gs[g]; best = g; }
      gmask |= 1u << best;
    }
    for (int i = 0; i < E_; ++i)
      if (!((gmask >> (i >> 3)) & 1u)) swb[i] = 0.0f;
    int sel[TOPK_];
    float ssum = 0.0f;
    for (int k = 0; k < TOPK_; ++k) {
      int best = 0; float bv = -1e30f;
      for (int i = 0; i < E_; ++i)
        if (swb[i] > bv) { bv = swb[i]; best = i; }
      swb[best] = -1e30f;
      sel[k] = best;
      ssum += sc[best];
    }
    float inv = RSCALE_ / (ssum + 1e-20f);
    for (int k = 0; k < TOPK_; ++k) {
      int ex  = sel[k];
      int pos = atomicAdd(&counts[ex], 1);
      tok[(size_t)ex * T_ + pos] = t;
      wt[(size_t)ex * T_ + pos]  = sc[ex] * inv;
    }
  }
}

// ---------------------------------------------------------------------------
// Shared expert phase A: act = silu(x@ws1^T) * (x@ws3^T). All tiles via TDM,
// double-buffered (issue k+1, wait TENSORcnt<=3, barrier, compute k).
// ---------------------------------------------------------------------------
__global__ __launch_bounds__(256) void shared_up_gate(
    const u16* __restrict__ hs, const u16* __restrict__ w1b,
    const u16* __restrict__ w3b, u16* __restrict__ act) {
  extern __shared__ __align__(16) char smem[];
  u16* aslab = (u16*)smem;                 // [2][128][72]
  u16* bslab = aslab + 2 * 128 * 72;       // [4][64][72]
  const int tid  = threadIdx.x;
  const int wave = tid >> 5, lane = tid & 31;
  const int col  = lane & 15, hi8 = (lane >> 4) << 3;
  const int t0 = blockIdx.x * 128;
  const int n0 = blockIdx.y * 64;

  v8f acc1[4], acc3[4];
  const v8f vzero = {0.f,0.f,0.f,0.f,0.f,0.f,0.f,0.f};
#pragma unroll
  for (int j = 0; j < 4; ++j) { acc1[j] = vzero; acc3[j] = vzero; }

  const u16* asrc  = hs  + (size_t)t0 * H_;
  const u16* b1src = w1b + (size_t)n0 * H_;
  const u16* b3src = w3b + (size_t)n0 * H_;

  if (wave == 0) {
    tdm_load_tile_2d(aslab,             asrc,  128, H_);
    tdm_load_tile_2d(bslab,             b1src, 64,  H_);
    tdm_load_tile_2d(bslab + 64 * 72,   b3src, 64,  H_);
  }
  for (int kc = 0; kc < H_ / 64; ++kc) {
    const int cur = kc & 1, nxt = cur ^ 1;
    if (wave == 0) {
      if (kc + 1 < H_ / 64) {
        tdm_load_tile_2d(aslab + nxt * 128 * 72,          asrc  + (kc + 1) * 64, 128, H_);
        tdm_load_tile_2d(bslab + (nxt * 2) * 64 * 72,     b1src + (kc + 1) * 64, 64,  H_);
        tdm_load_tile_2d(bslab + (nxt * 2 + 1) * 64 * 72, b3src + (kc + 1) * 64, 64,  H_);
        __builtin_amdgcn_s_wait_tensorcnt(3);
      } else {
        __builtin_amdgcn_s_wait_tensorcnt(0);
      }
    }
    __syncthreads();
    const u16* A  = aslab + cur * 128 * 72;
    const u16* B1 = bslab + (cur * 2) * 64 * 72;
    const u16* B3 = bslab + (cur * 2 + 1) * 64 * 72;
#pragma unroll
    for (int ks = 0; ks < 2; ++ks) {
      v16bf a = load_frag_a(A + wave * 16 * 72 + ks * 32, 72);
      v16bf bb1[4], bb3[4];
#pragma unroll
      for (int j = 0; j < 4; ++j) {
        bb1[j] = load_frag_b(B1 + j * 16 * 72 + ks * 32, 72);
        bb3[j] = load_frag_b(B3 + j * 16 * 72 + ks * 32, 72);
      }
#pragma unroll
      for (int j = 0; j < 4; ++j) {
        acc1[j] = wmma_bf16(a, bb1[j], acc1[j]);
        acc3[j] = wmma_bf16(a, bb3[j], acc3[j]);
      }
    }
    __syncthreads();
  }
#pragma unroll
  for (int j = 0; j < 4; ++j)
#pragma unroll
    for (int r = 0; r < 8; ++r) {
      int m = t0 + wave * 16 + hi8 + r;
      int n = n0 + j * 16 + col;
      float v = silu(acc1[j][r]) * acc3[j][r];
      act[(size_t)m * IS_ + n] = f32_to_bf16(v);
    }
}

// ---------------------------------------------------------------------------
// Shared expert phase B: out = act @ ws2^T  (plain store initializes d_out)
// ---------------------------------------------------------------------------
__global__ __launch_bounds__(256) void shared_down(
    const u16* __restrict__ act, const u16* __restrict__ w2b,
    float* __restrict__ out) {
  extern __shared__ __align__(16) char smem[];
  u16* aslab = (u16*)smem;                 // [2][128][72]
  u16* bslab = aslab + 2 * 128 * 72;       // [2][64][72]
  const int tid  = threadIdx.x;
  const int wave = tid >> 5, lane = tid & 31;
  const int col  = lane & 15, hi8 = (lane >> 4) << 3;
  const int t0 = blockIdx.x * 128;
  const int n0 = blockIdx.y * 64;

  v8f acc[4];
  const v8f vzero = {0.f,0.f,0.f,0.f,0.f,0.f,0.f,0.f};
#pragma unroll
  for (int j = 0; j < 4; ++j) acc[j] = vzero;

  const u16* asrc = act + (size_t)t0 * IS_;
  const u16* bsrc = w2b + (size_t)n0 * IS_;

  if (wave == 0) {
    tdm_load_tile_2d(aslab, asrc, 128, IS_);
    tdm_load_tile_2d(bslab, bsrc, 64,  IS_);
  }
  for (int kc = 0; kc < IS_ / 64; ++kc) {
    const int cur = kc & 1, nxt = cur ^ 1;
    if (wave == 0) {
      if (kc + 1 < IS_ / 64) {
        tdm_load_tile_2d(aslab + nxt * 128 * 72, asrc + (kc + 1) * 64, 128, IS_);
        tdm_load_tile_2d(bslab + nxt * 64 * 72,  bsrc + (kc + 1) * 64, 64,  IS_);
        __builtin_amdgcn_s_wait_tensorcnt(2);
      } else {
        __builtin_amdgcn_s_wait_tensorcnt(0);
      }
    }
    __syncthreads();
    const u16* A = aslab + cur * 128 * 72;
    const u16* B = bslab + cur * 64 * 72;
#pragma unroll
    for (int ks = 0; ks < 2; ++ks) {
      v16bf a = load_frag_a(A + wave * 16 * 72 + ks * 32, 72);
      v16bf bb[4];
#pragma unroll
      for (int j = 0; j < 4; ++j) bb[j] = load_frag_b(B + j * 16 * 72 + ks * 32, 72);
#pragma unroll
      for (int j = 0; j < 4; ++j) acc[j] = wmma_bf16(a, bb[j], acc[j]);
    }
    __syncthreads();
  }
#pragma unroll
  for (int j = 0; j < 4; ++j)
#pragma unroll
    for (int r = 0; r < 8; ++r) {
      int m = t0 + wave * 16 + hi8 + r;
      int n = n0 + j * 16 + col;
      out[(size_t)m * H_ + n] = acc[j][r];
    }
}

// ---------------------------------------------------------------------------
// Fused routed-expert kernel, per (expert, 128-token tile).
// Phase 1: h = silu(x@w1^T)*(x@w3^T)*cw kept in LDS (gathered A via registers
//          + prefetch; w1/w3 tiles via double-buffered TDM).
// Phase 2: out += h @ w2^T (w2 tiles via double-buffered TDM, f32 atomics).
// ---------------------------------------------------------------------------
__global__ __launch_bounds__(256) void expert_fused(
    const u16* __restrict__ hs, const u16* __restrict__ w1b,
    const u16* __restrict__ w2b, const u16* __restrict__ w3b,
    const int* __restrict__ counts, const int* __restrict__ tok,
    const float* __restrict__ twt, float* __restrict__ out) {
  extern __shared__ __align__(16) char smem[];
  u16* hbuf   = (u16*)smem;                  // [128][520]
  u16* astage = hbuf + 128 * 520;            // [128][72]
  u16* bslab  = astage + 128 * 72;           // [4][64][72]
  int*   tids = (int*)(bslab + 4 * 64 * 72);
  float* wts  = (float*)(tids + 128);

  const int e   = blockIdx.y;
  const int cnt = counts[e];
  const int t0  = blockIdx.x * 128;
  if (t0 >= cnt) return;
  const int rows = min(128, cnt - t0);

  const int tid  = threadIdx.x;
  const int wave = tid >> 5, lane = tid & 31;
  const int col  = lane & 15, hi8 = (lane >> 4) << 3;

  const u16* w1p = w1b + (size_t)e * I_ * H_;
  const u16* w3p = w3b + (size_t)e * I_ * H_;
  const u16* w2p = w2b + (size_t)e * H_ * I_;

  for (int i = tid; i < 128; i += 256) {
    int valid = (i < rows);
    tids[i] = valid ? tok[(size_t)e * T_ + t0 + i] : -1;
    wts[i]  = valid ? twt[(size_t)e * T_ + t0 + i] : 0.0f;
  }
  __syncthreads();

  const v8f vzero = {0.f,0.f,0.f,0.f,0.f,0.f,0.f,0.f};

  // ---- Phase 1 ----
  for (int nc = 0; nc < I_ / 64; ++nc) {
    v8f acc1[4], acc3[4];
#pragma unroll
    for (int j = 0; j < 4; ++j) { acc1[j] = vzero; acc3[j] = vzero; }
    const u16* b1src = w1p + (size_t)(nc * 64) * H_;
    const u16* b3src = w3p + (size_t)(nc * 64) * H_;
    if (wave == 0) {
      tdm_load_tile_2d(bslab,           b1src, 64, H_);
      tdm_load_tile_2d(bslab + 64 * 72, b3src, 64, H_);
    }
    for (int kc = 0; kc < H_ / 64; ++kc) {
      const int cur = kc & 1, nxt = cur ^ 1;
      // gathered A tile: registers -> LDS (+ prefetch of next chunk)
      for (int idx = tid; idx < 128 * 8; idx += 256) {
        int r = idx >> 3, c = idx & 7;
        int tk = tids[r];
        uint4 v; v.x = v.y = v.z = v.w = 0u;
        if (tk >= 0) {
          const u16* src = hs + (size_t)tk * H_ + kc * 64 + c * 8;
          v = *(const uint4*)src;
          if (kc + 1 < H_ / 64) __builtin_prefetch(src + 64, 0, 0);
        }
        *(uint4*)(astage + r * 72 + c * 8) = v;
      }
      if (wave == 0) {
        if (kc + 1 < H_ / 64) {
          tdm_load_tile_2d(bslab + (nxt * 2) * 64 * 72,     b1src + (kc + 1) * 64, 64, H_);
          tdm_load_tile_2d(bslab + (nxt * 2 + 1) * 64 * 72, b3src + (kc + 1) * 64, 64, H_);
          __builtin_amdgcn_s_wait_tensorcnt(2);
        } else {
          __builtin_amdgcn_s_wait_tensorcnt(0);
        }
      }
      __syncthreads();
      const u16* B1 = bslab + (cur * 2) * 64 * 72;
      const u16* B3 = bslab + (cur * 2 + 1) * 64 * 72;
#pragma unroll
      for (int ks = 0; ks < 2; ++ks) {
        v16bf a = load_frag_a(astage + wave * 16 * 72 + ks * 32, 72);
        v16bf bb1[4], bb3[4];
#pragma unroll
        for (int j = 0; j < 4; ++j) {
          bb1[j] = load_frag_b(B1 + j * 16 * 72 + ks * 32, 72);
          bb3[j] = load_frag_b(B3 + j * 16 * 72 + ks * 32, 72);
        }
#pragma unroll
        for (int j = 0; j < 4; ++j) {
          acc1[j] = wmma_bf16(a, bb1[j], acc1[j]);
          acc3[j] = wmma_bf16(a, bb3[j], acc3[j]);
        }
      }
      __syncthreads();
    }
#pragma unroll
    for (int j = 0; j < 4; ++j)
#pragma unroll
      for (int r = 0; r < 8; ++r) {
        int m = wave * 16 + hi8 + r;
        int n = nc * 64 + j * 16 + col;
        float v = silu(acc1[j][r]) * acc3[j][r] * wts[m];
        hbuf[m * 520 + n] = f32_to_bf16(v);
      }
    __syncthreads();
  }

  // ---- Phase 2 ----
  for (int hc = 0; hc < H_ / 64; ++hc) {
    v8f acc[4];
#pragma unroll
    for (int j = 0; j < 4; ++j) acc[j] = vzero;
    const u16* bsrc = w2p + (size_t)(hc * 64) * I_;
    if (wave == 0) tdm_load_tile_2d(bslab, bsrc, 64, I_);
    for (int kc = 0; kc < I_ / 64; ++kc) {
      const int cur = kc & 1, nxt = cur ^ 1;
      if (wave == 0) {
        if (kc + 1 < I_ / 64) {
          tdm_load_tile_2d(bslab + nxt * 64 * 72, bsrc + (kc + 1) * 64, 64, I_);
          __builtin_amdgcn_s_wait_tensorcnt(1);
        } else {
          __builtin_amdgcn_s_wait_tensorcnt(0);
        }
      }
      __syncthreads();
      const u16* B = bslab + cur * 64 * 72;
#pragma unroll
      for (int ks = 0; ks < 2; ++ks) {
        v16bf a = load_frag_a(hbuf + wave * 16 * 520 + kc * 64 + ks * 32, 520);
        v16bf bb[4];
#pragma unroll
        for (int j = 0; j < 4; ++j) bb[j] = load_frag_b(B + j * 16 * 72 + ks * 32, 72);
#pragma unroll
        for (int j = 0; j < 4; ++j) acc[j] = wmma_bf16(a, bb[j], acc[j]);
      }
      __syncthreads();
    }
#pragma unroll
    for (int j = 0; j < 4; ++j)
#pragma unroll
      for (int r = 0; r < 8; ++r) {
        int m = wave * 16 + hi8 + r;
        if (m < rows) {
          int tkn = tids[m];
          atomicAdd(&out[(size_t)tkn * H_ + hc * 64 + j * 16 + col], acc[j][r]);
        }
      }
  }
}

// ---------------------------------------------------------------------------
// Host launcher
// ---------------------------------------------------------------------------
extern "C" void kernel_launch(void* const* d_in, const int* in_sizes, int n_in,
                              void* d_out, int out_size, void* d_ws, size_t ws_size,
                              hipStream_t stream) {
  (void)in_sizes; (void)n_in; (void)out_size; (void)ws_size;
  const float* hs_f  = (const float*)d_in[0];
  const float* gw_f  = (const float*)d_in[1];
  const float* eb_f  = (const float*)d_in[2];
  const float* w1_f  = (const float*)d_in[3];
  const float* w2_f  = (const float*)d_in[4];
  const float* w3_f  = (const float*)d_in[5];
  const float* ws1_f = (const float*)d_in[6];
  const float* ws2_f = (const float*)d_in[7];
  const float* ws3_f = (const float*)d_in[8];
  float* out = (float*)d_out;
  char*  ws  = (char*)d_ws;

  size_t off = 0;
  auto carve = [&](size_t bytes) -> char* {
    char* p = ws + off;
    off = (off + bytes + 255) & ~(size_t)255;
    return p;
  };
  u16* hs_b  = (u16*)carve((size_t)T_ * H_ * 2);
  u16* w1_b  = (u16*)carve((size_t)E_ * I_ * H_ * 2);
  u16* w3_b  = (u16*)carve((size_t)E_ * I_ * H_ * 2);
  u16* w2_b  = (u16*)carve((size_t)E_ * H_ * I_ * 2);
  u16* ws1_b = (u16*)carve((size_t)IS_ * H_ * 2);
  u16* ws3_b = (u16*)carve((size_t)IS_ * H_ * 2);
  u16* ws2_b = (u16*)carve((size_t)H_ * IS_ * 2);
  u16* act_b = (u16*)carve((size_t)T_ * IS_ * 2);
  int*   cnts = (int*)carve(E_ * sizeof(int));
  int*   tokl = (int*)carve((size_t)E_ * T_ * sizeof(int));
  float* wtl  = (float*)carve((size_t)E_ * T_ * sizeof(float));

  auto cvt = [&](const float* src, u16* dst, size_t n) {
    int n4 = (int)(n / 4);
    cvt_f32_bf16<<<(n4 + 255) / 256, 256, 0, stream>>>(src, (uint2*)dst, n4);
  };
  cvt(hs_f,  hs_b,  (size_t)T_ * H_);
  cvt(w1_f,  w1_b,  (size_t)E_ * I_ * H_);
  cvt(w3_f,  w3_b,  (size_t)E_ * I_ * H_);
  cvt(w2_f,  w2_b,  (size_t)E_ * H_ * I_);
  cvt(ws1_f, ws1_b, (size_t)IS_ * H_);
  cvt(ws3_f, ws3_b, (size_t)IS_ * H_);
  cvt(ws2_f, ws2_b, (size_t)H_ * IS_);

  hipMemsetAsync(cnts, 0, E_ * sizeof(int), stream);

  router_kernel<<<T_, E_, 0, stream>>>(hs_f, gw_f, eb_f, cnts, tokl, wtl);

  const size_t SMEM_UP = (size_t)(2 * 128 * 72 + 4 * 64 * 72) * 2;       // 73,728
  shared_up_gate<<<dim3(T_ / 128, IS_ / 64), 256, SMEM_UP, stream>>>(
      hs_b, ws1_b, ws3_b, act_b);

  const size_t SMEM_DN = (size_t)(2 * 128 * 72 + 2 * 64 * 72) * 2;       // 55,296
  shared_down<<<dim3(T_ / 128, H_ / 64), 256, SMEM_DN, stream>>>(
      act_b, ws2_b, out);

  const size_t SMEM_EX =
      (size_t)(128 * 520 + 128 * 72 + 4 * 64 * 72) * 2 + 128 * 8;        // 189,440
  expert_fused<<<dim3(T_ / 128, E_), 256, SMEM_EX, stream>>>(
      hs_b, w1_b, w2_b, w3_b, cnts, tokl, wtl, out);
}
